// GCN_75917841924788
// MI455X (gfx1250) — matrix operand order repaired and checked by reading, not compiled, not run on hardware
//
#include <hip/hip_runtime.h>

typedef __attribute__((ext_vector_type(16))) __bf16 v16bf;
typedef __attribute__((ext_vector_type(8)))  float  v8f;

#define NNODES 8192
#define HDIM   64
#define NGRAPH 128

// ---------- bf16 helpers (manual, storage as unsigned short) ----------
__device__ __forceinline__ float bf2f(unsigned int u16) {
    union { unsigned int i; float f; } v; v.i = (u16 & 0xFFFFu) << 16; return v.f;
}
__device__ __forceinline__ unsigned short f2bf(float f) {
    union { float f; unsigned int i; } v; v.f = f;
    unsigned int u = v.i;
    unsigned int r = (u + 0x7FFFu + ((u >> 16) & 1u)) >> 16;   // RNE
    return (unsigned short)r;
}

// ---------- 1) zero dense A (128 MiB of bf16) ----------
__global__ void k_zero(uint4* p, size_t n16) {
    size_t i = (size_t)blockIdx.x * blockDim.x + threadIdx.x;
    size_t st = (size_t)gridDim.x * blockDim.x;
    uint4 z; z.x = z.y = z.z = z.w = 0u;
    for (; i < n16; i += st) p[i] = z;
}

// ---------- 2) scatter edges: A[r][c] = bf16(1.0) (dup-collapse) ----------
__global__ void k_scatter(unsigned short* A, const int* ei, int E) {
    int e = blockIdx.x * blockDim.x + threadIdx.x;
    if (e < E) {
        int r = ei[e];
        int c = ei[E + e];
        A[(size_t)r * NNODES + c] = 0x3F80u;   // 1.0 bf16
    }
}

// ---------- 3) row sums -> dinv = rsqrt(sum + 1) ----------
__global__ void k_rowsum(const unsigned short* A, float* dinv) {
    __shared__ float red[256];
    int r = blockIdx.x;
    const uint4* row = (const uint4*)(A + (size_t)r * NNODES);
    float s = 0.f;
    for (int i = threadIdx.x; i < NNODES / 8; i += 256) {
        uint4 q = row[i];
        s += bf2f(q.x) + bf2f(q.x >> 16) + bf2f(q.y) + bf2f(q.y >> 16)
           + bf2f(q.z) + bf2f(q.z >> 16) + bf2f(q.w) + bf2f(q.w >> 16);
    }
    red[threadIdx.x] = s; __syncthreads();
    for (int off = 128; off > 0; off >>= 1) {
        if (threadIdx.x < off) red[threadIdx.x] += red[threadIdx.x + off];
        __syncthreads();
    }
    if (threadIdx.x == 0) dinv[r] = rsqrtf(red[0] + 1.0f);
}

// ---------- 4) normalize in place: A = (A + I) * d_r * d_c ----------
__global__ void k_norm(unsigned short* A, const float* __restrict__ dinv) {
    size_t idx8 = ((size_t)blockIdx.x * blockDim.x + threadIdx.x) * 8;
    int r  = (int)(idx8 >> 13);          // / 8192
    int c0 = (int)(idx8 & (NNODES - 1));
    float dr = dinv[r];
    uint4* p = (uint4*)(A + idx8);
    uint4 q = *p;
    unsigned int e[8] = { q.x & 0xFFFFu, q.x >> 16, q.y & 0xFFFFu, q.y >> 16,
                          q.z & 0xFFFFu, q.z >> 16, q.w & 0xFFFFu, q.w >> 16 };
    unsigned int o[8];
#pragma unroll
    for (int i = 0; i < 8; ++i) {
        float v = bf2f(e[i]) + ((c0 + i) == r ? 1.0f : 0.0f);
        v *= dr * dinv[c0 + i];
        o[i] = f2bf(v);
    }
    q.x = o[0] | (o[1] << 16); q.y = o[2] | (o[3] << 16);
    q.z = o[4] | (o[5] << 16); q.w = o[6] | (o[7] << 16);
    *p = q;
}

// ---------- 5) BT[h][n] = bf16( x[n]@theta1 )  (F_IN = 7) ----------
__global__ void k_xtheta(const float* __restrict__ x, const float* __restrict__ th,
                         unsigned short* __restrict__ BT) {
    int t = blockIdx.x * blockDim.x + threadIdx.x;   // < 8192*64
    int n = t >> 6, h = t & 63;
    float s = 0.f;
#pragma unroll
    for (int f = 0; f < 7; ++f) s += x[n * 7 + f] * th[f * HDIM + h];
    BT[(size_t)h * NNODES + n] = f2bf(s);
}

// ---------- 6) BT[h2][n] = bf16( Hf[n]@theta )  (64x64) ----------
__global__ void k_htheta(const float* __restrict__ Hf, const float* __restrict__ th,
                         unsigned short* __restrict__ BT) {
    int t = blockIdx.x * blockDim.x + threadIdx.x;
    int n = t >> 6, h2 = t & 63;
    float s = 0.f;
#pragma unroll 8
    for (int h = 0; h < HDIM; ++h) s += Hf[n * HDIM + h] * th[h * HDIM + h2];
    BT[(size_t)h2 * NNODES + n] = f2bf(s);
}

// ---------- 7) WMMA GEMM: Hf (8192x64, f32) = [relu]( A(8192x8192 bf16) @ B ) ----------
// A row-major; B supplied transposed: BT (64 x 8192) bf16, row-major.
// Per wave: two 16-row M-tiles x four 16-col N-tiles (32x64 output, 8 accumulators).
// Register ping-pong double buffering: fragments for k+32 load while k computes.
union BFfrag { uint4 q[2]; v16bf v; };

__device__ __forceinline__ void ldfragA(BFfrag& d, const unsigned short* p) {
    d.q[0] = *(const uint4*)p;          // K 0..7   (lanes 0-15) / 8..15  (lanes 16-31)
    d.q[1] = *(const uint4*)(p + 16);   // K 16..23 (lanes 0-15) / 24..31 (lanes 16-31)
}
__device__ __forceinline__ void ldfragB(BFfrag& d, const unsigned short* p) {
    d.q[0] = *(const uint4*)p;          // K 0..7 / 8..15 along BT row
    d.q[1] = *(const uint4*)(p + 16);
}

// f[0],f[1] = A-frags (m-tile 0/1); f[2..5] = B-frags (n-tile 0..3)
__device__ __forceinline__ void load_frags(BFfrag* f, const unsigned short* aB0,
                                           const unsigned short* aB1,
                                           const unsigned short* bB, int k) {
    ldfragA(f[0], aB0 + k);
    ldfragA(f[1], aB1 + k);
#pragma unroll
    for (int nt = 0; nt < 4; ++nt)
        ldfragB(f[2 + nt], bB + (size_t)nt * 16 * NNODES + k);
}

__device__ __forceinline__ void do_wmma(v8f* acc, const BFfrag* f) {
#pragma unroll
    for (int nt = 0; nt < 4; ++nt) {
        acc[nt]     = __builtin_amdgcn_wmma_f32_16x16x32_bf16(
            false, f[0].v, false, f[2 + nt].v, (short)0, acc[nt],     false, false);
        acc[4 + nt] = __builtin_amdgcn_wmma_f32_16x16x32_bf16(
            false, f[1].v, false, f[2 + nt].v, (short)0, acc[4 + nt], false, false);
    }
}

__global__ __launch_bounds__(128) void k_gemm(const unsigned short* __restrict__ A,
                                              const unsigned short* __restrict__ BT,
                                              float* __restrict__ H, int relu_flag) {
    const int K    = NNODES;
    const int lane = threadIdx.x & 31;
    const int wave = threadIdx.x >> 5;               // 0..3
    const int m0   = (blockIdx.x * 8 + wave * 2) * 16;   // two 16-row tiles: m0, m0+16

    v8f acc[8];
#pragma unroll
    for (int i = 0; i < 8; ++i) acc[i] = (v8f)(0.0f);

    // A-fragment (16-bit 16x32 layout): lanes 0-15 row=lane K 0..7/16..23,
    // lanes 16-31 same rows K 8..15/24..31 -> two contiguous b128 loads.
    const int rA    = lane & 15;
    const int koffA = (lane >> 4) * 8;
    const unsigned short* aB0 = A + (size_t)(m0 + rA) * K + koffA;
    const unsigned short* aB1 = aB0 + (size_t)16 * K;

    // B-fragment: lane = output column (N), K 0..7/16..23 (lanes 0-15), 8..15/24..31 (lanes 16-31).
    const int colB  = lane & 15;
    const int koffB = (lane >> 4) * 8;
    const unsigned short* bB = BT + (size_t)colB * K + koffB;

    BFfrag fr[2][6];
    load_frags(fr[0], aB0, aB1, bB, 0);

    for (int k0 = 0; k0 < K; k0 += 64) {
        // prefetch the streaming A operand a few K-steps ahead (global_prefetch_b8)
        __builtin_prefetch(aB0 + ((k0 + 1024) & (K - 1)), 0, 1);
        __builtin_prefetch(aB1 + ((k0 + 1024) & (K - 1)), 0, 1);

        load_frags(fr[1], aB0, aB1, bB, k0 + 32);          // prefetch k+32
        do_wmma(acc, fr[0]);                               // compute k
        load_frags(fr[0], aB0, aB1, bB, (k0 + 64) & (K - 1)); // prefetch k+64 (wraps on tail)
        do_wmma(acc, fr[1]);                               // compute k+32
    }

    // C/D layout: VGPR j -> M = j + (lane>>4)*8 ; N = (lane&15) + nt*16
    const int rowb = m0 + (lane >> 4) * 8;
    const int col  = lane & 15;
#pragma unroll
    for (int mt = 0; mt < 2; ++mt)
#pragma unroll
        for (int nt = 0; nt < 4; ++nt)
#pragma unroll
            for (int j = 0; j < 8; ++j) {
                float v = acc[mt * 4 + nt][j];
                if (relu_flag) v = fmaxf(v, 0.0f);
                H[(size_t)(rowb + mt * 16 + j) * HDIM + nt * 16 + col] = v;
            }
}

// ---------- 8) deterministic segment mean-pool + relu (batch is sorted) ----------
__global__ void k_pool(const float* __restrict__ Hf, const int* __restrict__ batch,
                       float* __restrict__ gbuf) {
    int t = blockIdx.x * blockDim.x + threadIdx.x;   // < 128*64
    if (t >= NGRAPH * HDIM) return;
    int g = t >> 6, h = t & 63;
    int lo = 0, hi = NNODES;
    while (lo < hi) { int m = (lo + hi) >> 1; if (batch[m] <  g) lo = m + 1; else hi = m; }
    int s0 = lo; hi = NNODES;
    while (lo < hi) { int m = (lo + hi) >> 1; if (batch[m] <= g) lo = m + 1; else hi = m; }
    int s1 = lo;
    float s = 0.f;
    for (int n = s0; n < s1; ++n) s += Hf[(size_t)n * HDIM + h];
    float mean = s / fmaxf((float)(s1 - s0), 1.0f);
    gbuf[g * HDIM + h] = fmaxf(mean, 0.0f);
}

// ---------- 9) head: out = g @ W + b  (128x64 @ 64x2) ----------
__global__ void k_head(const float* __restrict__ gbuf, const float* __restrict__ W,
                       const float* __restrict__ b, float* __restrict__ out) {
    int t = threadIdx.x;                 // 256 threads
    int g = t >> 1, c = t & 1;
    float s = b[c];
#pragma unroll 8
    for (int h = 0; h < HDIM; ++h) s += gbuf[g * HDIM + h] * W[h * 2 + c];
    out[g * 2 + c] = s;
}

extern "C" void kernel_launch(void* const* d_in, const int* in_sizes, int n_in,
                              void* d_out, int out_size, void* d_ws, size_t ws_size,
                              hipStream_t stream) {
    const float* x     = (const float*)d_in[0];
    const int*   ei    = (const int*)  d_in[1];   // (2,E) int32
    const int*   batch = (const int*)  d_in[2];
    const float* th1   = (const float*)d_in[3];
    const float* th2   = (const float*)d_in[4];
    const float* th3   = (const float*)d_in[5];
    const float* W     = (const float*)d_in[6];
    const float* b     = (const float*)d_in[7];
    float*       out   = (float*)d_out;
    const int E = in_sizes[1] / 2;

    // workspace layout
    const size_t A_BYTES = (size_t)NNODES * NNODES * 2;        // 128 MiB bf16
    char* ws = (char*)d_ws;
    unsigned short* A    = (unsigned short*)ws;
    float*          dinv = (float*)(ws + A_BYTES);                              // 32 KiB
    unsigned short* BT   = (unsigned short*)(ws + A_BYTES + 32768);             // 1 MiB
    float*          Hf   = (float*)(ws + A_BYTES + 32768 + 1048576);            // 2 MiB
    float*          gbuf = (float*)(ws + A_BYTES + 32768 + 1048576 + 2097152);  // 32 KiB

    // build normalized dense adjacency in bf16 (fits MI455X 192MB L2)
    k_zero   <<<8192, 256, 0, stream>>>((uint4*)A, A_BYTES / 16);
    k_scatter<<<(E + 255) / 256, 256, 0, stream>>>(A, ei, E);
    k_rowsum <<<NNODES, 256, 0, stream>>>(A, dinv);
    k_norm   <<<(NNODES * (NNODES / 8)) / 256, 256, 0, stream>>>(A, dinv);

    // layer 1
    k_xtheta <<<(NNODES * HDIM) / 256, 256, 0, stream>>>(x, th1, BT);
    k_gemm   <<<NNODES / 128, 128, 0, stream>>>(A, BT, Hf, 1);
    // layer 2
    k_htheta <<<(NNODES * HDIM) / 256, 256, 0, stream>>>(Hf, th2, BT);
    k_gemm   <<<NNODES / 128, 128, 0, stream>>>(A, BT, Hf, 1);
    // layer 3 (no relu)
    k_htheta <<<(NNODES * HDIM) / 256, 256, 0, stream>>>(Hf, th3, BT);
    k_gemm   <<<NNODES / 128, 128, 0, stream>>>(A, BT, Hf, 0);

    // pool + head
    k_pool <<<(NGRAPH * HDIM + 255) / 256, 256, 0, stream>>>(Hf, batch, gbuf);
    k_head <<<1, 256, 0, stream>>>(gbuf, W, b, out);
}
